// ConnectionC2G_22299470201306
// MI455X (gfx1250) — matrix operand order repaired and checked by reading, not compiled, not run on hardware
//
#include <hip/hip_runtime.h>
#include <hip/hip_bf16.h>
#include <math.h>

// ---------------------------------------------------------------------------
// Problem constants (from the reference): B=8, N=4096, L=64*64=4096, C=256, GC=32
// ---------------------------------------------------------------------------
#define B_   8
#define N_   4096
#define L_   4096
#define C_   256
#define GC_  32
#define QSCALE 0.17677669529663687f   // 1/sqrt(32)
#define NSPLIT 4                       // waves per workgroup (split-K factor)

typedef __attribute__((ext_vector_type(16))) __bf16 v16bf;
typedef __attribute__((ext_vector_type(8)))  __bf16 v8bf;
typedef __attribute__((ext_vector_type(8)))  float  v8f;

__device__ __forceinline__ v8f wmma_bf16(v16bf a, v16bf b, v8f c) {
  // D(16x16 f32) = A(16x32 bf16) * B(32x16 bf16) + C
  return __builtin_amdgcn_wmma_f32_16x16x32_bf16(
      /*neg_a=*/false, a, /*neg_b=*/false, b,
      /*c_mod=*/(short)0, c, /*reuse_a=*/false, /*reuse_b=*/false);
}

__device__ __forceinline__ v16bf combine16(v8bf lo, v8bf hi) {
  v16bf r;
#pragma unroll
  for (int i = 0; i < 8; ++i) { r[i] = lo[i]; r[8 + i] = hi[i]; }
  return r;
}

__device__ __forceinline__ __bf16 to_bf(float x) { return (__bf16)x; }

// ---------------------------------------------------------------------------
// Kernel 1: K/V 1x1-conv projections -> bf16, WMMA-friendly layouts.
//   Kt  [B][L][32]  : Kt[l][k]  = (Wk @ img)[k][l] + bk[k]    (k contiguous)
//   Vrow[B][32][L]  : Vrow[o][l]= (Wv @ img)[o][l] + bv[o]    (l contiguous)
// grid: (L/256, 32, B), block: 256
// ---------------------------------------------------------------------------
__global__ __launch_bounds__(256) void proj_kv_kernel(
    const float* __restrict__ img, const float* __restrict__ Wk,
    const float* __restrict__ bk, const float* __restrict__ Wv,
    const float* __restrict__ bv, __bf16* __restrict__ Kt,
    __bf16* __restrict__ Vrow) {
  const int l = blockIdx.x * 256 + threadIdx.x;
  const int o = blockIdx.y;
  const int b = blockIdx.z;
  float accK = bk[o], accV = bv[o];
  const float* imgb = img + (size_t)b * C_ * L_ + l;
#pragma unroll 4
  for (int c = 0; c < C_; ++c) {
    float x = imgb[(size_t)c * L_];
    accK = fmaf(Wk[o * C_ + c], x, accK);
    accV = fmaf(Wv[o * C_ + c], x, accV);
  }
  Kt[((size_t)(b * L_ + l)) * GC_ + o]   = to_bf(accK);
  Vrow[((size_t)(b * GC_ + o)) * L_ + l] = to_bf(accV);
}

// ---------------------------------------------------------------------------
// Kernel 2: Q projection (scale 1/sqrt(32) folded in) -> bf16 Qbf[B][N][32]
// ---------------------------------------------------------------------------
__global__ __launch_bounds__(256) void proj_q_kernel(
    const float* __restrict__ graph, const float* __restrict__ Wq,
    const float* __restrict__ bq, __bf16* __restrict__ Qbf) {
  const int i = blockIdx.x * 256 + threadIdx.x;   // over B*N*32
  const int k = i & 31;
  const size_t bn = (size_t)(i >> 5);             // b*N + n
  float acc = bq[k];
  const float* g = graph + bn * GC_;
#pragma unroll
  for (int c = 0; c < GC_; ++c) acc = fmaf(Wq[k * GC_ + c], g[c], acc);
  Qbf[i] = to_bf(acc * QSCALE);
}

// ---------------------------------------------------------------------------
// Kernel 3 (pass A): per-(b,l) softmax stats over n (online max/sum).
// 4 waves per block; all share the same 16 l-columns, each wave streams a
// quarter of N in 16-row WMMA tiles; per-wave (m,Z) merged through LDS.
// grid: B * (L/16) blocks of 128 threads.
// ---------------------------------------------------------------------------
__global__ __launch_bounds__(32 * NSPLIT) void pass_a_kernel(
    const __bf16* __restrict__ Qbf, const __bf16* __restrict__ Kt,
    float* __restrict__ m_out, float* __restrict__ rz_out) {
  __shared__ float mbuf[NSPLIT][16];
  __shared__ float zbuf[NSPLIT][16];

  const int lane = threadIdx.x & 31;
  const int w    = threadIdx.x >> 5;   // wave id within block
  const int row  = lane & 15;
  const int half = lane >> 4;
  const int lt   = (blockIdx.x & (L_ / 16 - 1)) * 16;
  const int b    = blockIdx.x >> 8;    // L_/16 == 256

  // B-fragment: column l = lt+row, K = half*16 .. half*16+15 (contiguous)
  const __bf16* kp = Kt + ((size_t)(b * L_ + lt + row)) * GC_ + half * 16;
  const v16bf bfr = *(const v16bf*)kp;

  float m = -INFINITY, Z = 0.0f;
  const int nbeg = w * (N_ / NSPLIT), nend = nbeg + (N_ / NSPLIT);
  for (int n0 = nbeg; n0 < nend; n0 += 16) {
    // A-fragment: row n = n0+row; K = half*16 .. half*16+15 (contiguous)
    const __bf16* qp = Qbf + ((size_t)(b * N_ + n0 + row)) * GC_ + half * 16;
    v16bf a = combine16(*(const v8bf*)qp, *(const v8bf*)(qp + 8));
    v8f c = {};
    v8f d = wmma_bf16(a, bfr, c);   // d[r] = S[n0 + r + half*8, lt + row]

    float dmax = d[0];
#pragma unroll
    for (int r = 1; r < 8; ++r) dmax = fmaxf(dmax, d[r]);
    dmax = fmaxf(dmax, __shfl_xor(dmax, 16, 32));   // combine both row halves
    float mnew = fmaxf(m, dmax);
    float s = 0.0f;
#pragma unroll
    for (int r = 0; r < 8; ++r) s += __expf(d[r] - mnew);
    s += __shfl_xor(s, 16, 32);
    Z = Z * __expf(m - mnew) + s;
    m = mnew;
  }
  if (lane < 16) { mbuf[w][lane] = m; zbuf[w][lane] = Z; }
  __syncthreads();

  // merge the 4 per-wave partials: Z = sum_w Z_w * exp(m_w - m)
  if (threadIdx.x < 16) {
    const int t = threadIdx.x;
    float mm = mbuf[0][t];
#pragma unroll
    for (int j = 1; j < NSPLIT; ++j) mm = fmaxf(mm, mbuf[j][t]);
    float zz = 0.0f;
#pragma unroll
    for (int j = 0; j < NSPLIT; ++j) zz += zbuf[j][t] * __expf(mbuf[j][t] - mm);
    m_out[(size_t)b * L_ + lt + t]  = mm;
    rz_out[(size_t)b * L_ + lt + t] = 1.0f / zz;
  }
}

// ---------------------------------------------------------------------------
// Kernel 4 (pass B): recompute scores, normalize, and accumulate
//   msgT[n][o] = sum_l P[n,l] * V[o,l]
// 4 waves per block share one 16-row n-tile; each wave covers a quarter of L
// (split-K) with per-tile: 2 score WMMAs + LDS transpose of P + 2 message
// WMMAs. Partial 16x32 f32 tiles summed through LDS.
// grid: B*(N/16) blocks of 128 threads.
// ---------------------------------------------------------------------------
__global__ __launch_bounds__(32 * NSPLIT) void pass_b_kernel(
    const __bf16* __restrict__ Qbf, const __bf16* __restrict__ Kt,
    const __bf16* __restrict__ Vrow, const float* __restrict__ m_arr,
    const float* __restrict__ rz_arr, float* __restrict__ msgT) {
  __shared__ __attribute__((aligned(32))) __bf16 ptile[NSPLIT][16 * 32];
  __shared__ float red[NSPLIT][16][32];

  const int lane = threadIdx.x & 31;
  const int w    = threadIdx.x >> 5;
  const int row  = lane & 15;
  const int half = lane >> 4;
  const int nt   = (blockIdx.x & (N_ / 16 - 1)) * 16;
  const int b    = blockIdx.x >> 8;    // N_/16 == 256

  // Q A-fragment for this n-tile (fixed for whole loop)
  const __bf16* qp = Qbf + ((size_t)(b * N_ + nt + row)) * GC_ + half * 16;
  const v16bf aq = combine16(*(const v8bf*)qp, *(const v8bf*)(qp + 8));

  const float* mb  = m_arr  + (size_t)b * L_;
  const float* rzb = rz_arr + (size_t)b * L_;
  __bf16* pt = &ptile[w][0];

  v8f acc0 = {};   // msgT partial, o = 0..15
  v8f acc1 = {};   // msgT partial, o = 16..31

  const int lbeg = w * (L_ / NSPLIT), lend = lbeg + (L_ / NSPLIT);
  for (int lt = lbeg; lt < lend; lt += 32) {
    // ---- scores for l columns [lt, lt+32) : two 16x16 WMMAs ----
    const __bf16* kp0 = Kt + ((size_t)(b * L_ + lt + row)) * GC_ + half * 16;
    const v16bf bk0 = *(const v16bf*)kp0;
    const v16bf bk1 = *(const v16bf*)(kp0 + 16 * GC_);
    v8f z = {};
    v8f s0 = wmma_bf16(aq, bk0, z);   // cols lt + row
    v8f s1 = wmma_bf16(aq, bk1, z);   // cols lt + 16 + row

    // ---- normalize: P = exp(S - m[l]) / Z[l], store bf16 tile to LDS ----
    const float m0 = mb[lt + row],      rz0 = rzb[lt + row];
    const float m1 = mb[lt + 16 + row], rz1 = rzb[lt + 16 + row];
#pragma unroll
    for (int r = 0; r < 8; ++r) {
      const int nloc = r + half * 8;
      pt[nloc * 32 + row]      = to_bf(__expf(s0[r] - m0) * rz0);
      pt[nloc * 32 + 16 + row] = to_bf(__expf(s1[r] - m1) * rz1);
    }
    // same-wave LDS ops are in-order; compiler inserts s_wait_dscnt for reads

    // ---- re-read P as an A-fragment (row n, 32 l-values) ----
    const __bf16* pp = &pt[row * 32 + half * 16];
    const v16bf ap = combine16(*(const v8bf*)pp, *(const v8bf*)(pp + 8));

    // ---- V B-fragments: column o, K(l) = lt + half*16 + 0..15 contiguous ----
    const __bf16* vp0 = Vrow + ((size_t)(b * GC_ + row))      * L_ + lt + half * 16;
    const __bf16* vp1 = Vrow + ((size_t)(b * GC_ + 16 + row)) * L_ + lt + half * 16;
    const v16bf bv0 = *(const v16bf*)vp0;
    const v16bf bv1 = *(const v16bf*)vp1;

    acc0 = wmma_bf16(ap, bv0, acc0);
    acc1 = wmma_bf16(ap, bv1, acc1);
  }

  // ---- park partial tiles in LDS: acc0[r] -> (n_loc = r+half*8, o = row) ----
#pragma unroll
  for (int r = 0; r < 8; ++r) {
    red[w][r + half * 8][row]      = acc0[r];
    red[w][r + half * 8][16 + row] = acc1[r];
  }
  __syncthreads();

  // ---- sum the NSPLIT partials and write msgT[n][o] (f32) ----
  for (int j = threadIdx.x; j < 16 * 32; j += 32 * NSPLIT) {
    const int tr = j >> 5, tc = j & 31;
    float s = red[0][tr][tc];
#pragma unroll
    for (int k = 1; k < NSPLIT; ++k) s += red[k][tr][tc];
    msgT[((size_t)(b * N_ + nt + tr)) * GC_ + tc] = s;
  }
}

// ---------------------------------------------------------------------------
// Kernel 5: out = graph + Wc @ msgT + bc   (tiny 32x32 mix, f32 for accuracy)
// ---------------------------------------------------------------------------
__global__ __launch_bounds__(256) void final_kernel(
    const float* __restrict__ graph, const float* __restrict__ msgT,
    const float* __restrict__ Wc, const float* __restrict__ bc,
    float* __restrict__ out) {
  const int i = blockIdx.x * 256 + threadIdx.x;   // over B*N*32
  const int o = i & 31;
  const size_t bn = (size_t)(i >> 5);
  float acc = bc[o];
  const float* msg = msgT + bn * GC_;
#pragma unroll
  for (int c = 0; c < GC_; ++c) acc = fmaf(Wc[o * GC_ + c], msg[c], acc);
  out[i] = graph[i] + acc;
}

// ---------------------------------------------------------------------------
// Launch.  Workspace layout (bytes):
//   Qbf  bf16 [B*N*32]   @ 0         (2 MiB)
//   Kt   bf16 [B*L*32]   @ 2 MiB     (2 MiB)
//   Vrow bf16 [B*32*L]   @ 4 MiB     (2 MiB)
//   m    f32  [B*L]      @ 6 MiB     (128 KiB)
//   rz   f32  [B*L]      @ 6.125 MiB (128 KiB)
//   msgT f32  [B*N*32]   @ 6.25 MiB  (4 MiB)      total ~10.25 MiB
// ---------------------------------------------------------------------------
extern "C" void kernel_launch(void* const* d_in, const int* in_sizes, int n_in,
                              void* d_out, int out_size, void* d_ws, size_t ws_size,
                              hipStream_t stream) {
  const float* graph = (const float*)d_in[0];
  const float* img   = (const float*)d_in[1];
  const float* Wq    = (const float*)d_in[2];
  const float* bq    = (const float*)d_in[3];
  const float* Wk    = (const float*)d_in[4];
  const float* bk    = (const float*)d_in[5];
  const float* Wv    = (const float*)d_in[6];
  const float* bv    = (const float*)d_in[7];
  const float* Wc    = (const float*)d_in[8];
  const float* bc    = (const float*)d_in[9];
  float* out = (float*)d_out;

  char* ws = (char*)d_ws;
  const size_t MB = 1024 * 1024;
  __bf16* Qbf  = (__bf16*)(ws + 0);
  __bf16* Kt   = (__bf16*)(ws + 2 * MB);
  __bf16* Vrow = (__bf16*)(ws + 4 * MB);
  float*  m_s  = (float*)(ws + 6 * MB);
  float*  rz_s = (float*)(ws + 6 * MB + 128 * 1024);
  float*  msgT = (float*)(ws + 6 * MB + 256 * 1024);

  // 1) K/V projections -> bf16 staging
  proj_kv_kernel<<<dim3(L_ / 256, GC_, B_), 256, 0, stream>>>(
      img, Wk, bk, Wv, bv, Kt, Vrow);

  // 2) Q projection (scaled) -> bf16
  proj_q_kernel<<<(B_ * N_ * GC_) / 256, 256, 0, stream>>>(graph, Wq, bq, Qbf);

  // 3) softmax stats over n per (b,l)   (4-way split over n)
  pass_a_kernel<<<B_ * (L_ / 16), 32 * NSPLIT, 0, stream>>>(Qbf, Kt, m_s, rz_s);

  // 4) attention message: msgT[b][n][o] (4-way split over l)
  pass_b_kernel<<<B_ * (N_ / 16), 32 * NSPLIT, 0, stream>>>(
      Qbf, Kt, Vrow, m_s, rz_s, msgT);

  // 5) residual + Wc mix
  final_kernel<<<(B_ * N_ * GC_) / 256, 256, 0, stream>>>(graph, msgT, Wc, bc, out);
}
